// FusedFuzzyDeepNet_74217034875579
// MI455X (gfx1250) — matrix operand order repaired and checked by reading, compiled, not run on hardware
//
#include <hip/hip_runtime.h>
#include <hip/hip_fp16.h>

typedef __attribute__((ext_vector_type(16))) _Float16 v16h;
typedef __attribute__((ext_vector_type(8)))  _Float16 v8h;
typedef __attribute__((ext_vector_type(8)))  float    v8f;

#define Dk      512
#define NCLASS  10
#define MT      64                 // rows per workgroup (halves L2 weight traffic vs 32)
#define THREADS 512                // 16 waves, 4 per SIMD32
#define NWAVES  (THREADS / 32)
#define MTILES  (MT / 16)          // 4 row-tiles
#define NSLICES (NWAVES / MTILES)  // 4 N-slices; waves in a slice share B streams
#define LDST    520                // padded LDS row stride in halves (1040 B: 16B-aligned, bank-staggered)
#define BUFBYTES (MT * LDST * 2)   // 66560 B per activation buffer
#define SMEM_BYTES (3 * BUFBYTES)  // 199680 B dynamic LDS (<= 320 KB WGP budget)

// ---------------------------------------------------------------------------
// Repack f32 weights [K][Nreal] row-major into f16 WMMA B-fragment order:
//   out[((nt*KC + kc)*32 + lane)*16 + j] = w[ kc*32 + (lane>>4)*16 + j ][ nt*16 + (lane&15) ]
// so each lane's 16-half fragment chunk is one contiguous 32B load.
// Columns >= Nreal are zero-padded (used for w_out: 10 -> 16).
// ---------------------------------------------------------------------------
__global__ void pack_weights(const float* __restrict__ w, _Float16* __restrict__ out,
                             int K, int Nreal, int Npad) {
  const int KC = K >> 5;
  const int total = (Npad >> 4) * KC * 512;
  for (int idx = blockIdx.x * blockDim.x + threadIdx.x; idx < total;
       idx += gridDim.x * blockDim.x) {
    const int j    = idx & 15;
    const int lane = (idx >> 4) & 31;
    const int frag = idx >> 9;
    const int kc   = frag % KC;
    const int nt   = frag / KC;
    const int k = (kc << 5) + ((lane >> 4) << 4) + j;
    const int n = (nt << 4) + (lane & 15);
    const float v = (n < Nreal) ? w[(size_t)k * Nreal + n] : 0.0f;
    out[idx] = (_Float16)v;
  }
}

__device__ __forceinline__ float fast_rcp(float x) { return __builtin_amdgcn_rcpf(x); }

// ---------------------------------------------------------------------------
// One GEMM stage: C[MTxN] = act(A[MTxK] @ B[KxN] + bias).
// A rows in LDS (f16, stride LDST halves). K split over A0 (kc < KC0) and A1.
// B packed in fragment order in global (L2-resident).
// Wave mapping: mt = wid % MTILES, N-slice = wid / MTILES, so the MTILES waves
// of a slice read identical B streams (WGP$/L2 temporal reuse).
// NB = N-tiles register-blocked per A-fragment load; k-loop unrolled x4 to
// bound the load-batching window (no VGPR spills at 16 waves/block).
// ACT: 0 = fuzzy gaussian, 1 = sigmoid, 2 = relu(sigmoid), 3 = raw f32 logits
// ---------------------------------------------------------------------------
template <int ACT, int NB>
__device__ __forceinline__ void gemm_stage(
    const _Float16* __restrict__ A0, const _Float16* __restrict__ A1,
    int KC, int KC0,
    const _Float16* __restrict__ Bp, int NT,
    const float* __restrict__ bias,
    const float* __restrict__ mu, const float* __restrict__ sigma,
    _Float16* __restrict__ Oh, float* __restrict__ Of,
    int wid, int lane)
{
  const int mlane = lane & 15;
  const int khalf = lane >> 4;
  const int mt    = wid % MTILES;
  const int slice = wid / MTILES;
  const int perslice = (NT + NSLICES - 1) / NSLICES;
  const int ntlo = slice * perslice;
  const int nthi = (ntlo + perslice < NT) ? (ntlo + perslice) : NT;
  const v8f vzero = {0.f, 0.f, 0.f, 0.f, 0.f, 0.f, 0.f, 0.f};
  const int arow = (mt * 16 + mlane) * LDST + khalf * 16;

  for (int nt0 = ntlo; nt0 + NB <= nthi; nt0 += NB) {  // wave-uniform bounds
    v8f acc[NB];
#pragma unroll
    for (int i = 0; i < NB; ++i) acc[i] = vzero;

    const _Float16* bbase = Bp + ((size_t)nt0 * KC) * 512 + lane * 16;

    // K region 1: A0, kc in [0, KC0)
#pragma unroll 4
    for (int kc = 0; kc < KC0; ++kc) {
      const _Float16* ab = A0 + arow + (kc << 5);
      v8h alo = *(const v8h*)(ab);
      v8h ahi = *(const v8h*)(ab + 8);
      v16h a = __builtin_shufflevector(alo, ahi,
                                       0, 1, 2, 3, 4, 5, 6, 7,
                                       8, 9, 10, 11, 12, 13, 14, 15);
#pragma unroll
      for (int i = 0; i < NB; ++i) {
        v16h b = *(const v16h*)(bbase + ((size_t)i * KC + kc) * 512);
        acc[i] = __builtin_amdgcn_wmma_f32_16x16x32_f16(
            false, a, false, b, (short)0, acc[i], false, false);
      }
    }
    // K region 2: A1, kc in [KC0, KC) (only the fuse stage has KC > KC0)
#pragma unroll 4
    for (int kc = KC0; kc < KC; ++kc) {
      const _Float16* ab = A1 + arow + ((kc - KC0) << 5);
      v8h alo = *(const v8h*)(ab);
      v8h ahi = *(const v8h*)(ab + 8);
      v16h a = __builtin_shufflevector(alo, ahi,
                                       0, 1, 2, 3, 4, 5, 6, 7,
                                       8, 9, 10, 11, 12, 13, 14, 15);
#pragma unroll
      for (int i = 0; i < NB; ++i) {
        v16h b = *(const v16h*)(bbase + ((size_t)i * KC + kc) * 512);
        acc[i] = __builtin_amdgcn_wmma_f32_16x16x32_f16(
            false, a, false, b, (short)0, acc[i], false, false);
      }
    }
    // Epilogue: C layout -> acc[i][r] is (row mt*16 + r + 8*khalf, col (nt0+i)*16 + mlane)
#pragma unroll
    for (int i = 0; i < NB; ++i) {
      const int col = (nt0 + i) * 16 + mlane;
      const float bs = (ACT == 3) ? 0.0f : bias[col];
      float muv = 0.f, isg = 0.f;
      if (ACT == 0) { muv = mu[col]; const float s = sigma[col]; isg = fast_rcp(s * s); }
#pragma unroll
      for (int r = 0; r < 8; ++r) {
        const int mrow = mt * 16 + r + khalf * 8;
        const float x = acc[i][r] + bs;
        if (ACT == 3) {
          Of[mrow * 16 + mlane] = x;
        } else {
          float y;
          if (ACT == 0) { const float d = x - muv; y = __expf(-d * d * isg); }
          else          { y = fast_rcp(1.0f + __expf(-x)); if (ACT == 2) y = fmaxf(y, 0.0f); }
          Oh[mrow * LDST + col] = (_Float16)y;
        }
      }
    }
  }
}

// ---------------------------------------------------------------------------
// Fully fused network: one workgroup owns 64 rows, whole chain through LDS.
// ---------------------------------------------------------------------------
__global__ __launch_bounds__(THREADS, 1)
void fused_fuzzy_net(const float* __restrict__ input,
                     const _Float16* __restrict__ wfzp, const float* __restrict__ bfz,
                     const float* __restrict__ mu,   const float* __restrict__ sigma,
                     const _Float16* __restrict__ w1p, const float* __restrict__ b1,
                     const _Float16* __restrict__ w2p, const float* __restrict__ b2,
                     const _Float16* __restrict__ wfup, const float* __restrict__ bfu,
                     const _Float16* __restrict__ wop, const float* __restrict__ bout,
                     float* __restrict__ out)
{
  extern __shared__ char smem[];
  _Float16* bufA = (_Float16*)(smem);                 // X, then h2
  _Float16* bufB = (_Float16*)(smem + BUFBYTES);      // fuzz, then logits (f32 alias)
  _Float16* bufC = (_Float16*)(smem + 2 * BUFBYTES);  // h1, then fused
  float* logits = (float*)(smem + BUFBYTES);

  const int tid  = threadIdx.x;
  const int lane = tid & 31;
  const int wid  = tid >> 5;
  const size_t row0 = (size_t)blockIdx.x * MT;

  // Stage X: load 64x512 f32 rows -> f16 LDS tile (coalesced float4)
  const float4* in4 = (const float4*)(input + row0 * Dk);
  for (int i = tid; i < MT * Dk / 4; i += THREADS) {
    const float4 v = in4[i];
    const int m = i / (Dk / 4);
    const int k = (i % (Dk / 4)) * 4;
    _Float16* p = bufA + m * LDST + k;
    p[0] = (_Float16)v.x; p[1] = (_Float16)v.y;
    p[2] = (_Float16)v.z; p[3] = (_Float16)v.w;
  }
  __syncthreads();

  // fuzz = exp(-((X@Wf+bf)-mu)^2 / sigma^2)        -> bufB
  gemm_stage<0, 2>(bufA, bufA, 16, 16, wfzp, 32, bfz, mu, sigma, bufB, nullptr, wid, lane);
  // h1 = sigmoid(X@W1 + b1)                        -> bufC
  gemm_stage<1, 2>(bufA, bufA, 16, 16, w1p, 32, b1, nullptr, nullptr, bufC, nullptr, wid, lane);
  __syncthreads();
  // h2 = sigmoid(h1@W2 + b2)                       -> bufA (X dead)
  gemm_stage<1, 2>(bufC, bufC, 16, 16, w2p, 32, b2, nullptr, nullptr, bufA, nullptr, wid, lane);
  __syncthreads();
  // fused = relu(sigmoid([fuzz|h2]@Wfu + bfu))     -> bufC (h1 dead), K=1024
  gemm_stage<2, 2>(bufB, bufA, 32, 16, wfup, 32, bfu, nullptr, nullptr, bufC, nullptr, wid, lane);
  __syncthreads();
  // logits = fused @ Wout (f32)                    -> logits LDS (fuzz dead)
  gemm_stage<3, 1>(bufC, bufC, 16, 16, wop, 1, nullptr, nullptr, nullptr, nullptr, logits, wid, lane);
  __syncthreads();

  // log_softmax over 10 classes, one thread per row
  if (tid < MT) {
    float v[NCLASS];
    float mx = -3.4e38f;
    for (int c = 0; c < NCLASS; ++c) {
      v[c] = logits[tid * 16 + c] + bout[c];
      mx = fmaxf(mx, v[c]);
    }
    float s = 0.f;
    for (int c = 0; c < NCLASS; ++c) s += __expf(v[c] - mx);
    const float lse = mx + __logf(s);
    float* o = out + (row0 + tid) * NCLASS;
    for (int c = 0; c < NCLASS; ++c) o[c] = v[c] - lse;
  }
}

// ---------------------------------------------------------------------------
extern "C" void kernel_launch(void* const* d_in, const int* in_sizes, int n_in,
                              void* d_out, int out_size, void* d_ws, size_t ws_size,
                              hipStream_t stream) {
  const float* input  = (const float*)d_in[0];
  const float* w_fuzz = (const float*)d_in[1];
  const float* b_fuzz = (const float*)d_in[2];
  const float* mu     = (const float*)d_in[3];
  const float* sigma  = (const float*)d_in[4];
  const float* w1     = (const float*)d_in[5];
  const float* b1     = (const float*)d_in[6];
  const float* w2     = (const float*)d_in[7];
  const float* b2     = (const float*)d_in[8];
  const float* w_fuse = (const float*)d_in[9];
  const float* b_fuse = (const float*)d_in[10];
  const float* w_out  = (const float*)d_in[11];
  const float* b_out  = (const float*)d_in[12];
  float* out = (float*)d_out;

  // workspace: packed f16 weights (~2.64 MB total, L2-resident)
  _Float16* ws    = (_Float16*)d_ws;
  _Float16* wfz_p = ws;                       // 512*512
  _Float16* w1_p  = wfz_p + 512 * 512;
  _Float16* w2_p  = w1_p + 512 * 512;
  _Float16* wfu_p = w2_p + 512 * 512;         // 1024*512
  _Float16* wo_p  = wfu_p + 1024 * 512;       // 16 tiles * 512 = 8192

  pack_weights<<<256, 256, 0, stream>>>(w_fuzz, wfz_p, 512, 512, 512);
  pack_weights<<<256, 256, 0, stream>>>(w1,     w1_p,  512, 512, 512);
  pack_weights<<<256, 256, 0, stream>>>(w2,     w2_p,  512, 512, 512);
  pack_weights<<<512, 256, 0, stream>>>(w_fuse, wfu_p, 1024, 512, 512);
  pack_weights<<<32,  256, 0, stream>>>(w_out,  wo_p,  512, 10, 16);

  // allow ~195 KB dynamic LDS per workgroup (WGP supports up to 320 KB)
  (void)hipFuncSetAttribute((const void*)fused_fuzzy_net,
                            hipFuncAttributeMaxDynamicSharedMemorySize, SMEM_BYTES);

  const int blocks = 65536 / MT;  // 1024
  fused_fuzzy_net<<<blocks, THREADS, SMEM_BYTES, stream>>>(
      input, wfz_p, b_fuzz, mu, sigma, w1_p, b1, w2_p, b2,
      wfu_p, b_fuse, wo_p, b_out, out);
}
